// VectorBiasIPMP_77644418777478
// MI455X (gfx1250) — compile-verified, matrix operand order checked by reading, not run on hardware
//
#include <hip/hip_runtime.h>

typedef __attribute__((ext_vector_type(16))) __bf16 v16bf;
typedef __attribute__((ext_vector_type(8)))  float  v8f;

#define C_S 128
#define C_V 16
#define C_Z 64
#define C_H 256
#define K1  464   // 2*C_S + C_Z + 9*C_V
#define K1P 480   // padded to 15 chunks of 32
#define KC1 15
#define NT1 16    // C_H/16
#define KC2 8     // C_H/32
#define NT2 16
#define KC3 8
#define NT3 8     // C_S/16
#define MT  4     // M-tiles (of 16 edges) per workgroup
#define EPW (MT * 16)   // 64 edges per workgroup

// ---------------------------------------------------------------------------
// Zero the scalar-output accumulator and per-node counts (graph-replay safe).
// ---------------------------------------------------------------------------
__global__ void zero_buf(float* out_ss, unsigned* cnt, int N_) {
  int i = blockIdx.x * 256 + threadIdx.x;
  if (i < N_ * C_S) out_ss[i] = 0.0f;
  if (i < N_)       cnt[i] = 0u;
}

// ---------------------------------------------------------------------------
// Per-node precompute: s_pts -> rotate -> vecs (+trans/10), out_s_v, bf16 cache
// of node_scalars. ~0.2 GFLOP total; VALU is fine.
// ---------------------------------------------------------------------------
__global__ void node_prep(const float* __restrict__ ns, const float* __restrict__ nv,
                          const float* __restrict__ rot, const float* __restrict__ trans,
                          const float* __restrict__ W_pts, const float* __restrict__ b_pts,
                          const float* __restrict__ W_comb, const float* __restrict__ W_spts,
                          float* __restrict__ vecs, __bf16* __restrict__ nsbf,
                          float* __restrict__ out_sv, int N_) {
  int n = blockIdx.x * 64 + threadIdx.x;
  if (n >= N_) return;
  const float* s = ns + (size_t)n * C_S;
  for (int j = 0; j < C_S; ++j) nsbf[(size_t)n * C_S + j] = (__bf16)s[j];

  // s_pts_raw[o] = ns . W_pts[:,o] + b_pts[o], o = d*8 + p
  float raw[24];
  for (int o = 0; o < 24; ++o) {
    float acc = b_pts[o];
    for (int j = 0; j < C_S; ++j) acc += s[j] * W_pts[j * 24 + o];
    raw[o] = acc;
  }
  float R[3][3];
  for (int i = 0; i < 3; ++i)
    for (int j = 0; j < 3; ++j) R[i][j] = rot[(size_t)n * 9 + i * 3 + j];
  float tr[3];
  for (int d = 0; d < 3; ++d) tr[d] = trans[(size_t)n * 3 + d];

  // vecs_cat: 16 vector channels + 8 rotated scalar-derived points
  float cat[24][3];
  for (int c = 0; c < 16; ++c)
    for (int d = 0; d < 3; ++d) cat[c][d] = nv[((size_t)n * 16 + c) * 3 + d];
  for (int p = 0; p < 8; ++p)
    for (int i = 0; i < 3; ++i) {
      float a = 0.0f;
      for (int j = 0; j < 3; ++j) a += R[i][j] * raw[j * 8 + p];  // s_pts[n,p,j]=raw[j*8+p]
      cat[16 + p][i] = a;
    }
  for (int v = 0; v < 16; ++v) {
    float o0 = 0, o1 = 0, o2 = 0, q0 = 0, q1 = 0, q2 = 0;
    for (int c = 0; c < 24; ++c) {
      float w = W_comb[c * 16 + v], u = W_spts[c * 16 + v];
      o0 += cat[c][0] * w; o1 += cat[c][1] * w; o2 += cat[c][2] * w;
      q0 += cat[c][0] * u; q1 += cat[c][1] * u; q2 += cat[c][2] * u;
    }
    size_t bo = (size_t)n * 48 + v * 3;
    vecs[bo + 0] = o0 + tr[0] * 0.1f;
    vecs[bo + 1] = o1 + tr[1] * 0.1f;
    vecs[bo + 2] = o2 + tr[2] * 0.1f;
    out_sv[bo + 0] = q0; out_sv[bo + 1] = q1; out_sv[bo + 2] = q2;
  }
}

// ---------------------------------------------------------------------------
// Pack W1/W2/W3 into bf16 WMMA B fragments.
// Fragment (kc,nt): 32 lanes x 16 bf16 contiguous. Lane layout mirrors the
// documented 16-bit A layout: n = nt*16 + lane%16, half = lane/16,
// k(e) = kc*32 + (e>>3)*16 + half*8 + ((e>>1)&3)*2 + (e&1).
// ---------------------------------------------------------------------------
__global__ void pack_weights(const float* __restrict__ W1, const float* __restrict__ W2,
                             const float* __restrict__ W3,
                             __bf16* __restrict__ f1, __bf16* __restrict__ f2,
                             __bf16* __restrict__ f3) {
  int idx = blockIdx.x * 256 + threadIdx.x;
  const int t1 = KC1 * NT1 * 512;
  const int t2 = t1 + KC2 * NT2 * 512;
  const int t3 = t2 + KC3 * NT3 * 512;
  if (idx >= t3) return;
  const float* W; __bf16* F; int Ncols, Kact, rel, NT;
  if (idx < t1)      { W = W1; F = f1; Ncols = C_H; Kact = K1;  rel = idx;      NT = NT1; }
  else if (idx < t2) { W = W2; F = f2; Ncols = C_H; Kact = C_H; rel = idx - t1; NT = NT2; }
  else               { W = W3; F = f3; Ncols = C_S; Kact = C_H; rel = idx - t2; NT = NT3; }
  int frag   = rel >> 9;
  int within = rel & 511;
  int lane = within >> 4;
  int e    = within & 15;
  int kc = frag / NT, nt = frag % NT;
  int n = nt * 16 + (lane & 15);
  int half = lane >> 4;
  int v = e >> 1, hi = e & 1;
  int k = kc * 32 + ((v >> 2) << 4) + half * 8 + ((v & 3) << 1) + hi;
  float val = (k < Kact) ? W[(size_t)k * Ncols + n] : 0.0f;
  F[rel] = (__bf16)val;
}

// ---------------------------------------------------------------------------
__global__ void count_edges(const int* __restrict__ eidx, unsigned* __restrict__ cnt, int E_) {
  int e = blockIdx.x * 256 + threadIdx.x;
  if (e < E_) atomicAdd(&cnt[eidx[E_ + e]], 1u);
}

// ---------------------------------------------------------------------------
// Load a 16x32 bf16 A fragment (v16bf per lane) from an LDS row-major tile.
// ISA 16-bit A layout: lane m=lane&15, half=lane>>4;
// VGPR v holds K = (v>>2)*16 + half*8 + (v&3)*2 {+0,+1}.
// Lowers to two ds_load_b128 per lane (bytes [0..15] and [32..47] of the row
// window) — verified in round-1 disassembly.
// ---------------------------------------------------------------------------
__device__ __forceinline__ v16bf load_a_frag(const __bf16* buf, int stride, int kbase, int lane) {
  int m = lane & 15, half = lane >> 4;
  const __bf16* row = buf + m * stride + kbase + half * 8;
  v16bf a;
#pragma unroll
  for (int v = 0; v < 8; ++v) {
    int off = ((v >> 2) << 4) + ((v & 3) << 1);
    a[2 * v]     = row[off];
    a[2 * v + 1] = row[off + 1];
  }
  return a;
}

// ---------------------------------------------------------------------------
// Edge MLP: 64 edges (4 M-tiles) per 256-thread workgroup (8 wave32).
// B fragments are reused across 4 M-tiles (4x less L2 weight traffic), A
// fragments across 2 N-tiles. All K loops are FULLY UNROLLED so accumulators
// stay pinned in fixed VGPR ranges (no pipeliner register rotation / v_mov
// storms, no cross-iteration D->B WMMA hazards). LDS: the 64x512 m_in stage
// (64KB) is reused as the layer-2 output buffer after layer 1 consumes it.
// ---------------------------------------------------------------------------
__global__ void __launch_bounds__(256)
edge_mlp(const int* __restrict__ eidx, const float* __restrict__ ef,
         const float* __restrict__ rot, const float* __restrict__ trans,
         const float* __restrict__ vecs, const __bf16* __restrict__ nsbf,
         const __bf16* __restrict__ f1, const __bf16* __restrict__ f2,
         const __bf16* __restrict__ f3,
         const float* __restrict__ b1, const float* __restrict__ b2,
         const float* __restrict__ b3,
         float* __restrict__ out_ss, int E_) {
  __shared__ __bf16 sBig[EPW * 512];   // 64KB: m_in stage, later reused as H1 (stride 272)
  __shared__ __bf16 sH0[EPW * 272];    // 34KB: layer-1 output
  __shared__ int sSrc[EPW], sDst[EPW];

  const int t = threadIdx.x;
  const int base = blockIdx.x * EPW;

  if (t < EPW) {
    int e = base + t;
    if (e < E_) { sSrc[t] = eidx[e]; sDst[t] = eidx[E_ + e]; }
    else        { sSrc[t] = 0;       sDst[t] = -1; }
  }
  __syncthreads();

  // ---- Stage 1: build m_in tile (16 threads per edge, 4 edge groups) ----
#pragma unroll
  for (int g = 0; g < MT; ++g) {
    int el = g * 16 + (t >> 4);
    int ln = t & 15;
    int e = base + el;
    int draw = sDst[el];
    bool valid = draw >= 0;
    int dst = valid ? draw : 0;
    int src = sSrc[el];
    __bf16* row = &sBig[el * 512];
    // node scalars: dst then src (8 each per thread)
#pragma unroll
    for (int q = 0; q < 8; ++q) {
      row[ln * 8 + q]       = nsbf[(size_t)dst * C_S + ln * 8 + q];
      row[C_S + ln * 8 + q] = nsbf[(size_t)src * C_S + ln * 8 + q];
    }
    // edge features (4 per thread), fp32 -> bf16
#pragma unroll
    for (int q = 0; q < 4; ++q) {
      float x = valid ? ef[(size_t)e * C_Z + ln * 4 + q] : 0.0f;
      row[2 * C_S + ln * 4 + q] = (__bf16)x;
    }
    // local-frame points: thread ln handles vector channel c = ln
    {
      int c = ln;
      const float* R = &rot[(size_t)dst * 9];
      float t0 = trans[(size_t)dst * 3 + 0];
      float t1 = trans[(size_t)dst * 3 + 1];
      float t2 = trans[(size_t)dst * 3 + 2];
      size_t di = (size_t)dst * 48 + c * 3;
      size_t si = (size_t)src * 48 + c * 3;
      float vi0 = vecs[di + 0] - t0, vi1 = vecs[di + 1] - t1, vi2 = vecs[di + 2] - t2;
      float vj0 = vecs[si + 0] - t0, vj1 = vecs[si + 1] - t1, vj2 = vecs[si + 2] - t2;
      // loc[i] = sum_j R[j][i] * v[j]   (R^T v)
      float li0 = R[0] * vi0 + R[3] * vi1 + R[6] * vi2;
      float li1 = R[1] * vi0 + R[4] * vi1 + R[7] * vi2;
      float li2 = R[2] * vi0 + R[5] * vi1 + R[8] * vi2;
      float lj0 = R[0] * vj0 + R[3] * vj1 + R[6] * vj2;
      float lj1 = R[1] * vj0 + R[4] * vj1 + R[7] * vj2;
      float lj2 = R[2] * vj0 + R[5] * vj1 + R[8] * vj2;
      int o = 320 + c * 3;
      row[o + 0]      = (__bf16)li0; row[o + 1]      = (__bf16)li1; row[o + 2]      = (__bf16)li2;
      row[o + 48 + 0] = (__bf16)lj0; row[o + 48 + 1] = (__bf16)lj1; row[o + 48 + 2] = (__bf16)lj2;
      row[o + 96 + 0] = (__bf16)(lj0 - li0);
      row[o + 96 + 1] = (__bf16)(lj1 - li1);
      row[o + 96 + 2] = (__bf16)(lj2 - li2);
    }
    if (ln == 0)
      for (int k = K1; k < K1P; ++k) row[k] = (__bf16)0.0f;
  }
  __syncthreads();

  const int wave = t >> 5, lane = t & 31;
  const int col = lane & 15, half = lane >> 4;   // C/D: N = lane%16, M = r + 8*half
  const v16bf* F1 = (const v16bf*)f1;
  const v16bf* F2 = (const v16bf*)f2;
  const v16bf* F3 = (const v16bf*)f3;

  // ---- Layer 1: [64x480] x [480x256] -> relu -> sH0 ----
  {
    const int nt0 = wave * 2, nt1 = wave * 2 + 1;
    v8f acc[2][MT];
#pragma unroll
    for (int j = 0; j < 2; ++j)
#pragma unroll
      for (int mt = 0; mt < MT; ++mt) acc[j][mt] = (v8f){};
#pragma unroll
    for (int kc = 0; kc < KC1; ++kc) {
      v16bf b0 = F1[(kc * NT1 + nt0) * 32 + lane];
      v16bf b1f = F1[(kc * NT1 + nt1) * 32 + lane];
#pragma unroll
      for (int mt = 0; mt < MT; ++mt) {
        v16bf a = load_a_frag(&sBig[mt * 16 * 512], 512, kc * 32, lane);
        acc[0][mt] = __builtin_amdgcn_wmma_f32_16x16x32_bf16(false, a, false, b0,  (short)0, acc[0][mt], false, false);
        acc[1][mt] = __builtin_amdgcn_wmma_f32_16x16x32_bf16(false, a, false, b1f, (short)0, acc[1][mt], false, false);
      }
    }
#pragma unroll
    for (int j = 0; j < 2; ++j) {
      int nt = wave * 2 + j;
      float bias = b1[nt * 16 + col];
#pragma unroll
      for (int mt = 0; mt < MT; ++mt)
#pragma unroll
        for (int r = 0; r < 8; ++r) {
          float x = acc[j][mt][r] + bias;
          x = x > 0.0f ? x : 0.0f;
          sH0[(mt * 16 + r + 8 * half) * 272 + nt * 16 + col] = (__bf16)x;
        }
    }
  }
  __syncthreads();   // sBig fully consumed; safe to reuse as H1

  __bf16* sH1 = sBig;  // 64 rows x stride 272 (fits in 64x512 region)

  // ---- Layer 2: [64x256] x [256x256] -> relu -> sH1 (in sBig) ----
  {
    const int nt0 = wave * 2, nt1 = wave * 2 + 1;
    v8f acc[2][MT];
#pragma unroll
    for (int j = 0; j < 2; ++j)
#pragma unroll
      for (int mt = 0; mt < MT; ++mt) acc[j][mt] = (v8f){};
#pragma unroll
    for (int kc = 0; kc < KC2; ++kc) {
      v16bf b0 = F2[(kc * NT2 + nt0) * 32 + lane];
      v16bf b1f = F2[(kc * NT2 + nt1) * 32 + lane];
#pragma unroll
      for (int mt = 0; mt < MT; ++mt) {
        v16bf a = load_a_frag(&sH0[mt * 16 * 272], 272, kc * 32, lane);
        acc[0][mt] = __builtin_amdgcn_wmma_f32_16x16x32_bf16(false, a, false, b0,  (short)0, acc[0][mt], false, false);
        acc[1][mt] = __builtin_amdgcn_wmma_f32_16x16x32_bf16(false, a, false, b1f, (short)0, acc[1][mt], false, false);
      }
    }
#pragma unroll
    for (int j = 0; j < 2; ++j) {
      int nt = wave * 2 + j;
      float bias = b2[nt * 16 + col];
#pragma unroll
      for (int mt = 0; mt < MT; ++mt)
#pragma unroll
        for (int r = 0; r < 8; ++r) {
          float x = acc[j][mt][r] + bias;
          x = x > 0.0f ? x : 0.0f;
          sH1[(mt * 16 + r + 8 * half) * 272 + nt * 16 + col] = (__bf16)x;
        }
    }
  }
  __syncthreads();

  // ---- Layer 3: [64x256] x [256x128] -> atomic scatter-add to out_ss[dst] ----
  {
    int nt = wave;  // 8 waves, 8 N-tiles
    v8f acc[MT];
#pragma unroll
    for (int mt = 0; mt < MT; ++mt) acc[mt] = (v8f){};
#pragma unroll
    for (int kc = 0; kc < KC3; ++kc) {
      v16bf b = F3[(kc * NT3 + nt) * 32 + lane];
#pragma unroll
      for (int mt = 0; mt < MT; ++mt) {
        v16bf a = load_a_frag(&sH1[mt * 16 * 272], 272, kc * 32, lane);
        acc[mt] = __builtin_amdgcn_wmma_f32_16x16x32_bf16(false, a, false, b, (short)0, acc[mt], false, false);
      }
    }
    float bias = b3[nt * 16 + col];
#pragma unroll
    for (int mt = 0; mt < MT; ++mt)
#pragma unroll
      for (int r = 0; r < 8; ++r) {
        int m = mt * 16 + r + 8 * half;
        int d = sDst[m];
        if (d >= 0)
          atomicAdd(&out_ss[(size_t)d * C_S + nt * 16 + col], acc[mt][r] + bias);
      }
  }
}

// ---------------------------------------------------------------------------
__global__ void norm_kernel(float* __restrict__ out_ss, const unsigned* __restrict__ cnt, int N_) {
  int i = blockIdx.x * 256 + threadIdx.x;
  if (i >= N_ * C_S) return;
  unsigned c = cnt[i >> 7];
  float d = c > 0u ? (float)c : 1.0f;
  out_ss[i] /= d;
}

// ---------------------------------------------------------------------------
extern "C" void kernel_launch(void* const* d_in, const int* in_sizes, int n_in,
                              void* d_out, int out_size, void* d_ws, size_t ws_size,
                              hipStream_t stream) {
  const float* node_scalars  = (const float*)d_in[0];
  const float* node_vectors  = (const float*)d_in[1];
  const float* edge_features = (const float*)d_in[2];
  const int*   edge_index    = (const int*)d_in[3];
  const float* rot   = (const float*)d_in[4];
  const float* trans = (const float*)d_in[5];
  const float* W_pts = (const float*)d_in[6];
  const float* b_pts = (const float*)d_in[7];
  const float* W_comb = (const float*)d_in[8];
  const float* W_spts = (const float*)d_in[9];
  const float* W1 = (const float*)d_in[10];
  const float* b1 = (const float*)d_in[11];
  const float* W2 = (const float*)d_in[12];
  const float* b2 = (const float*)d_in[13];
  const float* W3 = (const float*)d_in[14];
  const float* b3 = (const float*)d_in[15];

  const int N_ = in_sizes[0] / C_S;
  const int E_ = in_sizes[3] / 2;

  float* out_ss = (float*)d_out;
  float* out_sv = out_ss + (size_t)N_ * C_S;

  // workspace carve-out
  char* w = (char*)d_ws;
  size_t off = 0;
  auto alignup = [](size_t x) { return (x + 255) & ~(size_t)255; };
  float*    vecs = (float*)(w + off);    off = alignup(off + (size_t)N_ * 48 * 4);
  __bf16*   nsbf = (__bf16*)(w + off);   off = alignup(off + (size_t)N_ * C_S * 2);
  unsigned* cnt  = (unsigned*)(w + off); off = alignup(off + (size_t)N_ * 4);
  __bf16*   f1   = (__bf16*)(w + off);   off = alignup(off + (size_t)KC1 * NT1 * 512 * 2);
  __bf16*   f2   = (__bf16*)(w + off);   off = alignup(off + (size_t)KC2 * NT2 * 512 * 2);
  __bf16*   f3   = (__bf16*)(w + off);   off = alignup(off + (size_t)KC3 * NT3 * 512 * 2);
  (void)off; (void)ws_size; (void)n_in; (void)out_size;

  hipLaunchKernelGGL(zero_buf, dim3((N_ * C_S + 255) / 256), dim3(256), 0, stream,
                     out_ss, cnt, N_);
  hipLaunchKernelGGL(node_prep, dim3((N_ + 63) / 64), dim3(64), 0, stream,
                     node_scalars, node_vectors, rot, trans, W_pts, b_pts, W_comb, W_spts,
                     vecs, nsbf, out_sv, N_);
  const int packTotal = (KC1 * NT1 + KC2 * NT2 + KC3 * NT3) * 512;
  hipLaunchKernelGGL(pack_weights, dim3((packTotal + 255) / 256), dim3(256), 0, stream,
                     W1, W2, W3, f1, f2, f3);
  hipLaunchKernelGGL(count_edges, dim3((E_ + 255) / 256), dim3(256), 0, stream,
                     edge_index, cnt, E_);
  hipLaunchKernelGGL(edge_mlp, dim3((E_ + EPW - 1) / EPW, 1, 1), dim3(256), 0, stream,
                     edge_index, edge_features, rot, trans, vecs, nsbf,
                     f1, f2, f3, b1, b2, b3, out_ss, E_);
  hipLaunchKernelGGL(norm_kernel, dim3((N_ * C_S + 255) / 256), dim3(256), 0, stream,
                     out_ss, cnt, N_);
}
